// QuantumKernelAttention_65481071398187
// MI455X (gfx1250) — matrix operand order repaired and checked by reading, not compiled, not run on hardware
//
#include <hip/hip_runtime.h>

typedef __attribute__((ext_vector_type(16))) _Float16 v16h;
typedef __attribute__((ext_vector_type(8)))  _Float16 v8h;
typedef __attribute__((ext_vector_type(8)))  float    v8f;
typedef __attribute__((ext_vector_type(4)))  float    v4f;

#define S_LEN 1024
#define N_W   4

// ---------------------------------------------------------------------------
// Kernel 1: per-token 16-dim feature vector
//   f_t = (c0,s0) (x) (c1,s1) (x) (c2,s2) (x) (c3,s3),  ck=cos(x_k/2), sk=sin(x_k/2)
// stored as f16 rows of 16 (32 bytes/token) in workspace. Thread 0 also writes
// a 32-byte zero pad row at F[n_tok] used by the attention kernel's K-padding
// lanes (makes the B-operand load exec-uniform).
// ---------------------------------------------------------------------------
__global__ void qk_features_65481071398187(const float* __restrict__ x,
                                           _Float16* __restrict__ F,
                                           int n_tok) {
  const int t = blockIdx.x * blockDim.x + threadIdx.x;
  if (t == 0) {                       // zero pad row for B-operand upper-K lanes
    v16h z = {};
    *(v16h*)(F + (size_t)n_tok * 16) = z;
  }
  if (t >= n_tok) return;
  const v4f xv = *(const v4f*)(x + (size_t)t * N_W);
  float c[4], s[4];
#pragma unroll
  for (int k = 0; k < 4; ++k) {
    const float a = 0.5f * xv[k];
    s[k] = __sinf(a);
    c[k] = __cosf(a);
  }
  float f2[2], f4[4], f8[8], f16[16];
  f2[0] = c[0]; f2[1] = s[0];
#pragma unroll
  for (int j = 0; j < 2; ++j) { f4[2 * j] = f2[j] * c[1]; f4[2 * j + 1] = f2[j] * s[1]; }
#pragma unroll
  for (int j = 0; j < 4; ++j) { f8[2 * j] = f4[j] * c[2]; f8[2 * j + 1] = f4[j] * s[2]; }
#pragma unroll
  for (int j = 0; j < 8; ++j) { f16[2 * j] = f8[j] * c[3]; f16[2 * j + 1] = f8[j] * s[3]; }
  v16h fv;
#pragma unroll
  for (int i = 0; i < 16; ++i) fv[i] = (_Float16)f16[i];
  *(v16h*)(F + (size_t)t * 16) = fv;
}

// ---------------------------------------------------------------------------
// Kernel 2: fused attention. One wave32 owns one 16-row i-tile of one batch.
// Streams 64 j-tiles: scores = A x B via v_wmma_f32_16x16x32_f16 (K=16 used,
// K=16..31 zero-padded), p = exp(|score|), accumulate l += p, o += p * x_j.
// Scores are in [0,1] so exp needs no max subtraction (matches softmax exactly).
// B-operand: lanes 0-15 stream F rows (col N=lane, K=0..15); lanes 16-31
// (K=16..31) read the stationary zero-pad row -> fully exec-uniform loop.
// ---------------------------------------------------------------------------
__global__ __launch_bounds__(256)
void qk_attention_65481071398187(const float* __restrict__ x,
                                 const _Float16* __restrict__ F,
                                 const _Float16* __restrict__ Fzero,
                                 float* __restrict__ out,
                                 int n_tiles) {
  const int wave = threadIdx.x >> 5;
  const int lane = threadIdx.x & 31;
  const int tile = blockIdx.x * (blockDim.x >> 5) + wave;
  if (tile >= n_tiles) return;

  const int b    = tile >> 6;          // S_LEN/16 = 64 tiles per batch
  const int i0   = (tile & 63) << 4;   // first row of this i-tile
  const int half = lane >> 4;          // 0: lanes 0-15, 1: lanes 16-31
  const int col  = lane & 15;
  const size_t batch_tok = (size_t)b * S_LEN;

  // ---- A operand (resident across the j-loop) ----
  // lane holds row M = col; half 0: elements 0..15 = F[row][0..15],
  // half 1 (K=16..31) = 0.
  union { v16h v; v8h h[2]; } au;
  v16h z16 = {};
  au.v = (half == 0) ? *(const v16h*)(F + (batch_tok + i0 + col) * 16) : z16;
  const v16h a = au.v;

  // ---- per-lane B pointer: data lanes walk F rows, pad lanes stay on zeros --
  const _Float16* bptr = (half == 0) ? (F + (batch_tok + col) * 16) : Fzero;
  const size_t    bstep = (half == 0) ? (size_t)16 * 16 : 0;  // halves per j-tile

  // ---- running softmax numerator/denominator (per lane = per column share) --
  float l[8];
  v4f   o[8];
  v4f   z4 = {};
#pragma unroll
  for (int r = 0; r < 8; ++r) { l[r] = 0.0f; o[r] = z4; }

#pragma unroll 2
  for (int j0 = 0; j0 < S_LEN; j0 += 16) {
    const v16h bv = *(const v16h*)bptr;
    bptr += bstep;
    const v4f xj = *(const v4f*)(x + (batch_tok + j0 + col) * N_W);

    v8f acc = {};
    acc = __builtin_amdgcn_wmma_f32_16x16x32_f16(
        /*neg_a=*/false, a, /*neg_b=*/false, bv,
        /*c_mod=*/(short)0, acc, /*reuse_a=*/false, /*reuse_b=*/false);

#pragma unroll
    for (int r = 0; r < 8; ++r) {     // register r -> row (r + 8*half), col = col
      const float p = __expf(fabsf(acc[r]));
      l[r] += p;
      o[r] += p * xj;
    }
  }

  // ---- reduce across the 16-lane column groups (butterfly, stays in-group) --
#pragma unroll
  for (int m = 1; m <= 8; m <<= 1) {
#pragma unroll
    for (int r = 0; r < 8; ++r) {
      l[r]   += __shfl_xor(l[r],   m, 32);
      o[r].x += __shfl_xor(o[r].x, m, 32);
      o[r].y += __shfl_xor(o[r].y, m, 32);
      o[r].z += __shfl_xor(o[r].z, m, 32);
      o[r].w += __shfl_xor(o[r].w, m, 32);
    }
  }

  // ---- write: lane with col<4 writes component `col` for its 8 rows ----
  if (col < N_W) {
#pragma unroll
    for (int r = 0; r < 8; ++r) {
      const int row = i0 + r + half * 8;
      const float num = (col == 0) ? o[r].x
                      : (col == 1) ? o[r].y
                      : (col == 2) ? o[r].z
                                   : o[r].w;
      out[(batch_tok + row) * N_W + col] = num / l[r];
    }
  }
}

// ---------------------------------------------------------------------------
extern "C" void kernel_launch(void* const* d_in, const int* in_sizes, int n_in,
                              void* d_out, int out_size, void* d_ws, size_t ws_size,
                              hipStream_t stream) {
  const float* x = (const float*)d_in[0];
  float* out = (float*)d_out;
  const int total = in_sizes[0];        // B * S * W
  const int n_tok = total / N_W;        // B * S
  _Float16* F = (_Float16*)d_ws;        // n_tok * 16 halves (+ 32B pad) ~ 1 MB
  const _Float16* Fzero = F + (size_t)n_tok * 16;

  qk_features_65481071398187<<<(n_tok + 255) / 256, 256, 0, stream>>>(x, F, n_tok);

  const int n_tiles = n_tok / 16;       // one wave per 16-row tile
  const int waves_per_block = 8;        // 256 threads = 8 wave32
  const int blocks = (n_tiles + waves_per_block - 1) / waves_per_block;
  qk_attention_65481071398187<<<blocks, 32 * waves_per_block, 0, stream>>>(
      x, F, Fzero, out, n_tiles);
}
